// SelfAttention_34230889349469
// MI455X (gfx1250) — compile-verified, matrix operand order checked by reading, not run on hardware
//
#include <hip/hip_runtime.h>
#include <cstdint>
#include <cstddef>

// Problem sizes (fixed by the reference)
constexpr int Bn = 4, Cn = 256, Ln = 2048, Hn = 8, En = 32;
constexpr float kScale = 0.0625f;         // 1/sqrt(C)
constexpr int HE = Hn * En;               // 256

typedef __attribute__((ext_vector_type(16))) __bf16 bf16x16;
typedef __attribute__((ext_vector_type(8)))  __bf16 bf16x8;
typedef __attribute__((ext_vector_type(8)))  float  f32x8;

// ---- helpers -------------------------------------------------------------

__device__ __forceinline__ __bf16 f2bf(float f) {
  unsigned u = __builtin_bit_cast(unsigned, f);
  unsigned r = (u + 0x7FFFu + ((u >> 16) & 1u)) >> 16;   // RNE
  unsigned short s = (unsigned short)r;
  return __builtin_bit_cast(__bf16, s);
}

// Load one 16x32 bf16 WMMA operand tile from a row-major [row][k] buffer.
// CDNA5 16-bit A/B layout: lane<16 holds row (row0+lane), K 0..7 & 16..23;
// lane>=16 holds row (row0+lane-16), K 8..15 & 24..31.  Two b128 loads/lane.
__device__ __forceinline__ bf16x16 load_tile(const __bf16* __restrict__ base,
                                             int row0, int stride, int k0) {
  const int lane = threadIdx.x & 31;
  const __bf16* p = base + (size_t)(row0 + (lane & 15)) * stride
                         + k0 + ((lane >> 4) << 3);
  bf16x8 lo = *reinterpret_cast<const bf16x8*>(p);
  bf16x8 hi = *reinterpret_cast<const bf16x8*>(p + 16);
  return __builtin_shufflevector(lo, hi, 0,1,2,3,4,5,6,7,8,9,10,11,12,13,14,15);
}

__device__ __forceinline__ f32x8 wmma_bf16(bf16x16 a, bf16x16 b, f32x8 c) {
  return __builtin_amdgcn_wmma_f32_16x16x32_bf16(false, a, false, b,
                                                 (short)0, c, false, false);
}

// ---- stage 0: casts / transposes ----------------------------------------

// x (b,c,l) f32 -> xbT (b,l,c) bf16
__global__ void cast_x_kernel(const float* __restrict__ x, __bf16* __restrict__ xbT) {
  int tid = blockIdx.x * blockDim.x + threadIdx.x;          // Bn*Cn*Ln
  int b = tid / (Cn * Ln);
  int rem = tid - b * (Cn * Ln);
  int c = rem / Ln;
  int l = rem - c * Ln;
  xbT[((size_t)b * Ln + l) * Cn + c] = f2bf(x[tid]);
}

// innerprod (h,k,q) f32 -> Wt (h,q,k) bf16
__global__ void cast_w_kernel(const float* __restrict__ W, __bf16* __restrict__ Wt) {
  int tid = blockIdx.x * blockDim.x + threadIdx.x;          // Hn*Cn*Cn
  int h = tid / (Cn * Cn);
  int rem = tid - h * (Cn * Cn);
  int k = rem / Cn;
  int q = rem - k * Cn;
  Wt[((size_t)h * Cn + q) * Cn + k] = f2bf(W[tid]);
}

// tovalue (h,t,c) f32 -> Tb same layout bf16
__global__ void cast_t_kernel(const float* __restrict__ tv, __bf16* __restrict__ Tb) {
  int tid = blockIdx.x * blockDim.x + threadIdx.x;          // Hn*En*Cn
  Tb[tid] = f2bf(tv[tid]);
}

// unifyheads (h,c,t) f32 -> Ub (c, h*En+t) bf16
__global__ void cast_u_kernel(const float* __restrict__ U, __bf16* __restrict__ Ub) {
  int tid = blockIdx.x * blockDim.x + threadIdx.x;          // Hn*Cn*En
  int h = tid / (Cn * En);
  int rem = tid - h * (Cn * En);
  int c = rem / En;
  int t = rem - c * En;
  Ub[(size_t)c * HE + h * En + t] = f2bf(U[tid]);
}

// ---- stage 1: xwT[b,h,i,q] = sum_k x[b,k,i] * W[h,k,q]  (bf16 GEMM) ------

__global__ void xw_gemm_kernel(const __bf16* __restrict__ xbT,
                               const __bf16* __restrict__ Wt,
                               __bf16* __restrict__ xwT) {
  int wid = blockIdx.x * 8 + (threadIdx.x >> 5);   // Bn*Hn*(Ln/16)*(Cn/16)=65536
  int qt  = wid & 15;
  int it  = (wid >> 4) & 127;
  int bh  = wid >> 11;
  int b   = bh >> 3, h = bh & 7;
  int i0 = it * 16, q0 = qt * 16;

  const __bf16* Abase = xbT + (size_t)b * Ln * Cn;    // rows=i, stride Cn
  const __bf16* Bbase = Wt + (size_t)h * Cn * Cn;     // rows=q, stride Cn

  f32x8 acc = {};
#pragma unroll
  for (int kc = 0; kc < Cn; kc += 32) {
    bf16x16 a = load_tile(Abase, i0, Cn, kc);
    bf16x16 bt = load_tile(Bbase, q0, Cn, kc);
    acc = wmma_bf16(a, bt, acc);
  }
  const int lane = threadIdx.x & 31;
  const int rb = (lane >> 4) << 3;
  const int q = q0 + (lane & 15);
#pragma unroll
  for (int r = 0; r < 8; ++r)
    xwT[((size_t)bh * Ln + (i0 + rb + r)) * Cn + q] = f2bf(acc[r]);
}

// ---- stage 2: v[b,h,t,i] = sum_c T[h,t,c] * x[b,c,i]  (bf16 GEMM) --------

__global__ void v_gemm_kernel(const __bf16* __restrict__ Tb,
                              const __bf16* __restrict__ xbT,
                              __bf16* __restrict__ vbuf) {
  int wid = blockIdx.x * 8 + (threadIdx.x >> 5);   // Bn*Hn*2*(Ln/16)=8192
  int tt  = wid & 1;
  int it  = (wid >> 1) & 127;
  int bh  = wid >> 8;
  int b   = bh >> 3, h = bh & 7;
  int t0 = tt * 16, i0 = it * 16;

  const __bf16* Abase = Tb + (size_t)h * En * Cn;     // rows=t, stride Cn
  const __bf16* Bbase = xbT + (size_t)b * Ln * Cn;    // rows=i, stride Cn

  f32x8 acc = {};
#pragma unroll
  for (int kc = 0; kc < Cn; kc += 32) {
    bf16x16 a = load_tile(Abase, t0, Cn, kc);
    bf16x16 bt = load_tile(Bbase, i0, Cn, kc);
    acc = wmma_bf16(a, bt, acc);
  }
  const int lane = threadIdx.x & 31;
  const int rb = (lane >> 4) << 3;
  const int i = i0 + (lane & 15);
#pragma unroll
  for (int r = 0; r < 8; ++r)
    vbuf[((size_t)bh * En + (t0 + rb + r)) * Ln + i] = f2bf(acc[r]);
}

// ---- stage 3: per-row softmax stats a_i = m_i + log(sum_j exp(s_ij - m_i))

__global__ void stats_kernel(const __bf16* __restrict__ xwT,
                             const __bf16* __restrict__ xbT,
                             float* __restrict__ abuf) {
  int wid = blockIdx.x * 8 + (threadIdx.x >> 5);   // Bn*Hn*(Ln/16)=4096
  int it  = wid & 127;
  int bh  = wid >> 7;
  int b   = bh >> 3;
  int i0 = it * 16;

  const __bf16* Abase = xwT + (size_t)bh * Ln * Cn;   // rows=i, stride Cn
  const __bf16* Bbase = xbT + (size_t)b * Ln * Cn;    // rows=j, stride Cn

  bf16x16 Areg[8];
#pragma unroll
  for (int kc = 0; kc < 8; ++kc) Areg[kc] = load_tile(Abase, i0, Cn, kc * 32);

  float m[8], Z[8];
#pragma unroll
  for (int r = 0; r < 8; ++r) { m[r] = -3.0e38f; Z[r] = 0.0f; }

  for (int j0 = 0; j0 < Ln; j0 += 16) {
    f32x8 S = {};
#pragma unroll
    for (int kc = 0; kc < 8; ++kc) {
      bf16x16 bt = load_tile(Bbase, j0, Cn, kc * 32);
      S = wmma_bf16(Areg[kc], bt, S);
    }
#pragma unroll
    for (int r = 0; r < 8; ++r) {
      float s = S[r] * kScale;
      float nm = fmaxf(m[r], s);
      Z[r] = Z[r] * __expf(m[r] - nm) + __expf(s - nm);
      m[r] = nm;
    }
  }
  // combine 16 column-lanes per half-wave (lanes differ only in j)
#pragma unroll
  for (int off = 1; off < 16; off <<= 1) {
#pragma unroll
    for (int r = 0; r < 8; ++r) {
      float om = __shfl_xor(m[r], off, 32);
      float oz = __shfl_xor(Z[r], off, 32);
      float nm = fmaxf(m[r], om);
      Z[r] = Z[r] * __expf(m[r] - nm) + oz * __expf(om - nm);
      m[r] = nm;
    }
  }
  const int lane = threadIdx.x & 31;
  if ((lane & 15) == 0) {
    const int rb = (lane >> 4) << 3;
#pragma unroll
    for (int r = 0; r < 8; ++r)
      abuf[(size_t)bh * Ln + i0 + rb + r] = m[r] + __logf(Z[r]);
  }
}

// ---- stage 4: y[t,j] = sum_i v[t,i] * exp(s_ij - a_i); ybuf (b, j, h*En+t)

__global__ void apply_kernel(const __bf16* __restrict__ xwT,
                             const __bf16* __restrict__ xbT,
                             const __bf16* __restrict__ vbuf,
                             const float* __restrict__ abuf,
                             const float* __restrict__ tvb,
                             __bf16* __restrict__ ybuf) {
  int wid = blockIdx.x * 8 + (threadIdx.x >> 5);   // Bn*Hn*(Ln/16)=4096
  int jt  = wid & 127;
  int bh  = wid >> 7;
  int b   = bh >> 3, h = bh & 7;
  int j0 = jt * 16;

  const __bf16* Abase = xwT + (size_t)bh * Ln * Cn;   // rows=i, stride Cn
  const __bf16* Bbase = xbT + (size_t)b * Ln * Cn;    // rows=j, stride Cn
  const __bf16* Vbase = vbuf + (size_t)bh * En * Ln;  // rows=t, stride Ln
  const float*  Arow  = abuf + (size_t)bh * Ln;

  const int lane = threadIdx.x & 31;
  const int rb = (lane >> 4) << 3;

  // B operand (this wave's 16 output columns) stays resident for the whole loop
  bf16x16 Breg[8];
#pragma unroll
  for (int kc = 0; kc < 8; ++kc) Breg[kc] = load_tile(Bbase, j0, Cn, kc * 32);

  f32x8 Y0 = {}, Y1 = {};

  for (int ic = 0; ic < Ln; ic += 32) {
    __builtin_prefetch(Abase + (size_t)(ic + 32) * Cn, 0, 1);
    f32x8 S0 = {}, S1 = {};
#pragma unroll
    for (int kc = 0; kc < 8; ++kc) {
      bf16x16 a0 = load_tile(Abase, ic, Cn, kc * 32);
      S0 = wmma_bf16(a0, Breg[kc], S0);
    }
#pragma unroll
    for (int kc = 0; kc < 8; ++kc) {
      bf16x16 a1 = load_tile(Abase, ic + 16, Cn, kc * 32);
      S1 = wmma_bf16(a1, Breg[kc], S1);
    }
    // P = exp(S*scale - a_i), repacked register-locally into a B operand:
    // elems 0..7 <- S0 rows, elems 8..15 <- S1 rows (K = i mod 32)
    bf16x16 P;
#pragma unroll
    for (int r = 0; r < 8; ++r) {
      float a0v = Arow[ic + rb + r];
      float a1v = Arow[ic + 16 + rb + r];
      P[r]     = f2bf(__expf(S0[r] * kScale - a0v));
      P[8 + r] = f2bf(__expf(S1[r] * kScale - a1v));
    }
    bf16x16 V0 = load_tile(Vbase, 0, Ln, ic);    // t rows 0..15,  K = i chunk
    bf16x16 V1 = load_tile(Vbase, 16, Ln, ic);   // t rows 16..31
    Y0 = wmma_bf16(V0, P, Y0);
    Y1 = wmma_bf16(V1, P, Y1);
  }

  // + tovalue_bias, emit bf16 ybuf[b][j][h*En + t] (contiguous t per lane)
  const int j = j0 + (lane & 15);
  __bf16* yrow = ybuf + ((size_t)b * Ln + j) * HE + h * En;
#pragma unroll
  for (int tt = 0; tt < 2; ++tt) {
    f32x8 Yt = tt ? Y1 : Y0;
    int tb = tt * 16 + rb;
    bf16x8 pack;
#pragma unroll
    for (int r = 0; r < 8; ++r) pack[r] = f2bf(Yt[r] + tvb[h * En + tb + r]);
    *reinterpret_cast<bf16x8*>(yrow + tb) = pack;
  }
}

// ---- stage 5: out[b,c,l] = sum_{ht} Ub[c,ht] * ybuf[b,l,ht] + bias[c] ----

__global__ void unify_kernel(const __bf16* __restrict__ Ub,
                             const __bf16* __restrict__ ybuf,
                             const float* __restrict__ ubias,
                             float* __restrict__ out) {
  int wid = blockIdx.x * 8 + (threadIdx.x >> 5);   // Bn*(Cn/16)*(Ln/16)=8192
  int lt  = wid & 127;
  int ct  = (wid >> 7) & 15;
  int b   = wid >> 11;
  int c0 = ct * 16, l0 = lt * 16;

  const __bf16* Bbase = ybuf + (size_t)b * Ln * HE;   // rows=l, stride HE

  f32x8 acc = {};
#pragma unroll
  for (int kc = 0; kc < HE; kc += 32) {
    bf16x16 a = load_tile(Ub, c0, HE, kc);
    bf16x16 bt = load_tile(Bbase, l0, HE, kc);
    acc = wmma_bf16(a, bt, acc);
  }
  const int lane = threadIdx.x & 31;
  const int rb = (lane >> 4) << 3;
  const int l = l0 + (lane & 15);
#pragma unroll
  for (int r = 0; r < 8; ++r) {
    int c = c0 + rb + r;
    out[((size_t)b * Cn + c) * Ln + l] = acc[r] + ubias[c];
  }
}

// ---- host launcher -------------------------------------------------------

extern "C" void kernel_launch(void* const* d_in, const int* in_sizes, int n_in,
                              void* d_out, int out_size, void* d_ws, size_t ws_size,
                              hipStream_t stream) {
  (void)in_sizes; (void)n_in; (void)out_size; (void)ws_size;

  const float* x   = (const float*)d_in[0];   // (B,C,L)
  const float* W   = (const float*)d_in[1];   // (H,C,C)
  const float* TV  = (const float*)d_in[2];   // (H,E,C)
  const float* TVB = (const float*)d_in[3];   // (1,H,E,1)
  const float* U   = (const float*)d_in[4];   // (H,C,E)
  const float* UB  = (const float*)d_in[5];   // (1,C,1)
  float* out = (float*)d_out;

  char* ws = (char*)d_ws;
  // workspace layout (bytes), all 16B-aligned
  __bf16* xbT  = (__bf16*)(ws + 0);           //  4 MiB  (b,l,c)
  __bf16* Wt   = (__bf16*)(ws + 4194304);     //  1 MiB  (h,q,k)
  __bf16* Tb   = (__bf16*)(ws + 5242880);     //  128KiB (h,t,c)
  __bf16* Ub   = (__bf16*)(ws + 5373952);     //  128KiB (c,ht)
  __bf16* xwT  = (__bf16*)(ws + 5505024);     // 32 MiB  (b,h,i,q)
  __bf16* vbuf = (__bf16*)(ws + 39059456);    //  4 MiB  (b,h,t,i)
  float*  abuf = (float* )(ws + 43253760);    // 256KiB  (b,h,i)
  __bf16* ybuf = (__bf16*)(ws + 43515904);    //  4 MiB  (b,l,ht)   total ~45.5 MiB

  // stage 0: casts / transposes
  cast_x_kernel<<<Bn * Cn * Ln / 256, 256, 0, stream>>>(x, xbT);
  cast_w_kernel<<<Hn * Cn * Cn / 256, 256, 0, stream>>>(W, Wt);
  cast_t_kernel<<<Hn * En * Cn / 256, 256, 0, stream>>>(TV, Tb);
  cast_u_kernel<<<Hn * Cn * En / 256, 256, 0, stream>>>(U, Ub);

  // stage 1-2: projection GEMMs
  xw_gemm_kernel<<<8192, 256, 0, stream>>>(xbT, Wt, xwT);
  v_gemm_kernel<<<1024, 256, 0, stream>>>(Tb, xbT, vbuf);

  // stage 3: softmax row statistics
  stats_kernel<<<512, 256, 0, stream>>>(xwT, xbT, abuf);

  // stage 4: fused scores + softmax + value aggregation
  apply_kernel<<<512, 256, 0, stream>>>(xwT, xbT, vbuf, abuf, TVB, ybuf);

  // stage 5: head unification
  unify_kernel<<<1024, 256, 0, stream>>>(Ub, ybuf, UB, out);
}
